// cross_attn_23029614641235
// MI455X (gfx1250) — compile-verified, matrix operand order checked by reading, not run on hardware
//
#include <hip/hip_runtime.h>
#include <hip/hip_bf16.h>

#define DEVINL __device__ __forceinline__

typedef __attribute__((ext_vector_type(16))) __bf16 v16bf;
typedef __attribute__((ext_vector_type(8)))  __bf16 v8bf;
typedef __attribute__((ext_vector_type(8)))  float  v8f;

static constexpr int NHEAD = 16, DK = 64, DV = 64, DMODEL = 1024;
static constexpr int BB = 2, LQ = 2048, LKK = 2048;
static constexpr int MTOK = BB * LQ;           // 4096 tokens
static constexpr int HD   = NHEAD * DV;        // 1024

// ---------------------------------------------------------------------------
// CDNA5 async global->LDS copy (ASYNCcnt-tracked). Each lane moves 16 B from
// its global address to its LDS byte offset. Generic pointers to __shared__
// carry the wave-relative LDS offset in their low 32 bits.
// ---------------------------------------------------------------------------
DEVINL unsigned lds_off_of(const void* p) {
    return (unsigned)(uintptr_t)p;
}
DEVINL void async_ld_b128(const void* gptr, unsigned lds_off) {
    asm volatile("global_load_async_to_lds_b128 %0, %1, off"
                 :: "v"(lds_off), "v"(gptr)
                 : "memory");
}
DEVINL void wait_async0() {
    asm volatile("s_wait_asynccnt 0" ::: "memory");
}

// ---------------------------------------------------------------------------
// Fragment loader: 16x32 bf16 A-fragment (or B-fragment from a transposed
// operand). Per ISA 7.12.2: lanes 0-15 hold K = {0..7} u {16..23} of row
// (lane&15); lanes 16-31 hold K = {8..15} u {24..31}.
// rowPtr must point at element [row][k0] with contiguous K (16-B aligned).
// ---------------------------------------------------------------------------
DEVINL v16bf load_frag16(const __bf16* rowPtr, int lane) {
    const int kOff = (lane >> 4) << 3;          // 0 or 8
    v8bf lo = *(const v8bf*)(rowPtr + kOff);
    v8bf hi = *(const v8bf*)(rowPtr + 16 + kOff);
    v16bf f;
#pragma unroll
    for (int i = 0; i < 8; ++i) { f[i] = lo[i]; f[i + 8] = hi[i]; }
    return f;
}

DEVINL v8f wmma_bf16(v16bf a, v16bf b, v8f c) {
    return __builtin_amdgcn_wmma_f32_16x16x32_bf16(false, a, false, b,
                                                   (short)0, c, false, false);
}

// ---------------------------------------------------------------------------
// Elementwise f32 -> bf16 cast (vectorized x4)
// ---------------------------------------------------------------------------
__global__ __launch_bounds__(256)
void cast_bf16_kernel(const float4* __restrict__ in, __bf16* __restrict__ out, int n4) {
    int i = blockIdx.x * 256 + threadIdx.x;
    if (i < n4) {
        float4 f = in[i];
        out[4 * i + 0] = (__bf16)f.x;
        out[4 * i + 1] = (__bf16)f.y;
        out[4 * i + 2] = (__bf16)f.z;
        out[4 * i + 3] = (__bf16)f.w;
    }
}

// ---------------------------------------------------------------------------
// Transpose + cast: w [K,N] f32 row-major -> wT [N,K] bf16 row-major
// ---------------------------------------------------------------------------
__global__ __launch_bounds__(256)
void transpose_cast_kernel(const float* __restrict__ w, __bf16* __restrict__ wT,
                           int K, int N) {
    __shared__ float tile[32][33];
    const int bx = blockIdx.x * 32;   // N
    const int by = blockIdx.y * 32;   // K
    const int tx = threadIdx.x & 31;
    const int ty = threadIdx.x >> 5;  // 8 rows / pass
#pragma unroll
    for (int r = ty; r < 32; r += 8)
        tile[r][tx] = w[(size_t)(by + r) * N + bx + tx];
    __syncthreads();
#pragma unroll
    for (int r = ty; r < 32; r += 8)
        wT[(size_t)(bx + r) * K + by + tx] = (__bf16)tile[tx][r];
}

// ---------------------------------------------------------------------------
// WMMA GEMM: C[M,N] = A[M,K](bf16,row-major) x BT[N,K](bf16,row-major == B^T)
// Block = 256 threads = 8 waves; each wave -> 16(M) x 64(N). The 64x32 B-tile
// is shared by all 8 waves: double-buffered in LDS via async global->LDS DMA.
// MODE 0: store bf16 head-major   [B,H,L,64]   (qh / kh)
// MODE 2: store bf16 head-major transposed [B,H,64,Lk]  (vhT)
// MODE 3: store f32  = acc + residual  (output projection)
// ---------------------------------------------------------------------------
static constexpr int BSTR = 40;   // B-tile LDS row stride (80 B: aligned, low-conflict)

template <int MODE>
__global__ __launch_bounds__(256)
void gemm_bf16_kernel(const __bf16* __restrict__ A, const __bf16* __restrict__ BT,
                      int M, int N, int K,
                      __bf16* __restrict__ obf, float* __restrict__ ofp,
                      const float* __restrict__ residual) {
    __shared__ __bf16 bLds[2][64][BSTR];

    const int tid   = threadIdx.x;
    const int lane  = tid & 31;
    const int wave  = tid >> 5;
    const int nBase = blockIdx.x * 64;
    const int mBase = blockIdx.y * 128 + wave * 16;
    const int nLane = lane & 15;

    // one 16-B chunk per thread: 64 rows x 4 chunks == 256 threads
    const int sRow = tid >> 2, sCol = (tid & 3) * 8;
    const __bf16* sGlob = BT + (size_t)(nBase + sRow) * K + sCol;

    auto stage = [&](int k0, int buf) {
        async_ld_b128(sGlob + k0, lds_off_of(&bLds[buf][sRow][sCol]));
    };

    stage(0, 0);

    const __bf16* aPtr = A + (size_t)(mBase + nLane) * K;
    v8f acc[4] = {};
    for (int k0 = 0; k0 < K; k0 += 32) {
        const int buf = (k0 >> 5) & 1;
        wait_async0();            // own async ops for buf done
        __syncthreads();          // everyone's done; safe to overwrite buf^1
        if (k0 + 32 < K) stage(k0 + 32, buf ^ 1);

        v16bf af = load_frag16(aPtr + k0, lane);
#pragma unroll
        for (int nt = 0; nt < 4; ++nt) {
            v16bf bf_ = load_frag16(&bLds[buf][nt * 16 + nLane][0], lane);
            acc[nt] = wmma_bf16(af, bf_, acc[nt]);
        }
    }

    const int mHalf = (lane >> 4) * 8;
#pragma unroll
    for (int nt = 0; nt < 4; ++nt) {
#pragma unroll
        for (int r = 0; r < 8; ++r) {
            const int m = mBase + mHalf + r;
            const int n = nBase + nt * 16 + nLane;
            const float val = acc[nt][r];
            if (MODE == 3) {
                ofp[(size_t)m * N + n] = val + residual[(size_t)m * N + n];
            } else {
                const int b = m >> 11, l = m & 2047;   // token -> (batch, pos)
                const int h = n >> 6,  d = n & 63;     // col   -> (head, dim)
                size_t idx;
                if (MODE == 2) idx = ((size_t)(b * NHEAD + h) * DV + d) * LKK + l;
                else           idx = ((size_t)(b * NHEAD + h) * LQ + l) * DK + d;
                obf[idx] = (__bf16)val;
            }
        }
    }
}

// ---------------------------------------------------------------------------
// Flash attention: grid (LQ/64, H, B), block 128 (4 waves).
// All 4 waves share one (b,h): the 32-key K/V chunk is staged once in LDS
// (async DMA, double-buffered) instead of 4x redundant global streams.
// ---------------------------------------------------------------------------
static constexpr int PSTR = 48;   // P   tile stride (96 B)
static constexpr int KSTR = 72;   // K   tile stride (144 B)
static constexpr int VSTR = 40;   // V^T tile stride (80 B)

__global__ __launch_bounds__(128)
void flash_attn_kernel(const __bf16* __restrict__ qh, const __bf16* __restrict__ kh,
                       const __bf16* __restrict__ vhT, const int* __restrict__ mask,
                       const float* __restrict__ mean, const float* __restrict__ stdv,
                       __bf16* __restrict__ attnout) {
    __shared__ __bf16 kLds[2][32][KSTR];   // 32 keys x 64 dims
    __shared__ __bf16 vLds[2][64][VSTR];   // 64 dims x 32 keys
    __shared__ __bf16 pLds[4][16][PSTR];   // per-wave P tile

    const int tid   = threadIdx.x;
    const int lane  = tid & 31;
    const int wave  = tid >> 5;
    const int qTile = blockIdx.x * 4 + wave;
    const int h     = blockIdx.y;
    const int b     = blockIdx.z;
    const int qBase = qTile * 16;
    const int nLane = lane & 15;
    const int mHalf = (lane >> 4) * 8;

    const __bf16* kBase = kh  + (size_t)(b * NHEAD + h) * LQ * DK;
    const __bf16* vBase = vhT + (size_t)(b * NHEAD + h) * DV * LKK;

    // staging geometry: K tile 32x64 (8 chunks/row), V tile 64x32 (4 chunks/row)
    const int kcA = tid, kcB = tid + 128;         // 256 chunks over 128 threads
    const int kRowA = kcA >> 3, kColA = (kcA & 7) * 8;
    const int kRowB = kcB >> 3, kColB = (kcB & 7) * 8;
    const int vRowA = kcA >> 2, vColA = (kcA & 3) * 8;
    const int vRowB = kcB >> 2, vColB = (kcB & 3) * 8;

    auto stage = [&](int k0, int buf) {
        async_ld_b128(kBase + (size_t)(k0 + kRowA) * DK + kColA,
                      lds_off_of(&kLds[buf][kRowA][kColA]));
        async_ld_b128(kBase + (size_t)(k0 + kRowB) * DK + kColB,
                      lds_off_of(&kLds[buf][kRowB][kColB]));
        async_ld_b128(vBase + (size_t)vRowA * LKK + k0 + vColA,
                      lds_off_of(&vLds[buf][vRowA][vColA]));
        async_ld_b128(vBase + (size_t)vRowB * LKK + k0 + vColB,
                      lds_off_of(&vLds[buf][vRowB][vColB]));
    };

    stage(0, 0);

    // Q fragments (reused over the whole K loop)
    const __bf16* qRow = qh + ((size_t)(b * NHEAD + h) * LQ + qBase + nLane) * DK;
    const v16bf qf0 = load_frag16(qRow, lane);        // d 0..31
    const v16bf qf1 = load_frag16(qRow + 32, lane);   // d 32..63

    const int*   maskB = mask + ((size_t)b * LQ + qBase) * LKK;
    const float* meanB = mean + (size_t)b * LKK;
    const float* stdB  = stdv + (size_t)b * LKK;

    v8f   oAcc[4] = {};
    float rowMax[8], rowSum[8];
#pragma unroll
    for (int r = 0; r < 8; ++r) { rowMax[r] = -1e30f; rowSum[r] = 0.f; }

    for (int k0 = 0; k0 < LKK; k0 += 32) {
        const int buf = (k0 >> 5) & 1;
        wait_async0();
        __syncthreads();
        if (k0 + 32 < LKK) stage(k0 + 32, buf ^ 1);

        // ---- scores: two 16x16 tiles over this 32-key chunk
        float sc[2][8];
#pragma unroll
        for (int t = 0; t < 2; ++t) {
            const int kcol = k0 + t * 16 + nLane;
            const __bf16* kRow = &kLds[buf][t * 16 + nLane][0];
            v16bf kf0 = load_frag16(kRow, lane);
            v16bf kf1 = load_frag16(kRow + 32, lane);
            v8f s = {};
            s = wmma_bf16(qf0, kf0, s);
            s = wmma_bf16(qf1, kf1, s);
            const float mn = meanB[kcol] * 0.125f;   // 1/sqrt(64) folded in
            const float sd = stdB[kcol]  * 0.125f;
#pragma unroll
            for (int r = 0; r < 8; ++r) {
                const int mk = maskB[(size_t)(mHalf + r) * LKK + kcol];
                sc[t][r] = mk ? (s[r] * mn + sd) : -1e9f;
            }
        }

        // ---- online softmax: row max across 16 lanes of each half-wave
        float scale[8];
#pragma unroll
        for (int r = 0; r < 8; ++r) {
            float m2 = fmaxf(sc[0][r], sc[1][r]);
#pragma unroll
            for (int off = 1; off < 16; off <<= 1)
                m2 = fmaxf(m2, __shfl_xor(m2, off, 32));
            const float nm = fmaxf(rowMax[r], m2);
            scale[r] = __expf(rowMax[r] - nm);
            rowMax[r] = nm;
        }

        // ---- P = exp(s - max): row sums + stage into LDS (wave-private)
        float ps[8];
#pragma unroll
        for (int r = 0; r < 8; ++r) ps[r] = 0.f;
#pragma unroll
        for (int t = 0; t < 2; ++t) {
#pragma unroll
            for (int r = 0; r < 8; ++r) {
                const float p = __expf(sc[t][r] - rowMax[r]);
                ps[r] += p;
                pLds[wave][mHalf + r][t * 16 + nLane] = (__bf16)p;
            }
        }
#pragma unroll
        for (int r = 0; r < 8; ++r) {
            float v2 = ps[r];
#pragma unroll
            for (int off = 1; off < 16; off <<= 1)
                v2 += __shfl_xor(v2, off, 32);
            rowSum[r] = rowSum[r] * scale[r] + v2;
        }

        // ---- rescale O, then O += P[16x32] x V[32x64]
#pragma unroll
        for (int nt = 0; nt < 4; ++nt)
#pragma unroll
            for (int r = 0; r < 8; ++r) oAcc[nt][r] *= scale[r];

        // wave-private LDS round trip: DS ops are in-order within a wave
        v16bf pf = load_frag16(&pLds[wave][nLane][0], lane);
#pragma unroll
        for (int nt = 0; nt < 4; ++nt) {
            v16bf vf = load_frag16(&vLds[buf][nt * 16 + nLane][0], lane);
            oAcc[nt] = wmma_bf16(pf, vf, oAcc[nt]);
        }
    }

    // ---- epilogue: O /= rowSum, store to [B, LQ, H*DV] bf16
    __bf16* outB = attnout + ((size_t)b * LQ + qBase) * HD + h * DV;
#pragma unroll
    for (int r = 0; r < 8; ++r) rowSum[r] = 1.0f / rowSum[r];
#pragma unroll
    for (int nt = 0; nt < 4; ++nt)
#pragma unroll
        for (int r = 0; r < 8; ++r)
            outB[(size_t)(mHalf + r) * HD + nt * 16 + nLane] =
                (__bf16)(oAcc[nt][r] * rowSum[r]);
}

// ---------------------------------------------------------------------------
// LayerNorm over 1024 cols; one block (256 thr) per row; single-pass moments.
// ---------------------------------------------------------------------------
__global__ __launch_bounds__(256)
void layernorm_kernel(const float* __restrict__ x, const float* __restrict__ gamma,
                      const float* __restrict__ beta, float* __restrict__ out) {
    __shared__ float redS[8], redQ[8];
    const int row = blockIdx.x, t = threadIdx.x;
    const float* xr = x + (size_t)row * DMODEL;
    float v[4], s = 0.f, sq = 0.f;
#pragma unroll
    for (int i = 0; i < 4; ++i) {
        float f = xr[t + 256 * i];
        v[i] = f; s += f; sq += f * f;
    }
#pragma unroll
    for (int off = 16; off > 0; off >>= 1) {
        s  += __shfl_xor(s,  off, 32);
        sq += __shfl_xor(sq, off, 32);
    }
    const int lane = t & 31, w = t >> 5;
    if (lane == 0) { redS[w] = s; redQ[w] = sq; }
    __syncthreads();
    if (w == 0) {
        float a = (lane < 8) ? redS[lane] : 0.f;
        float c = (lane < 8) ? redQ[lane] : 0.f;
#pragma unroll
        for (int off = 4; off > 0; off >>= 1) {
            a += __shfl_xor(a, off, 32);
            c += __shfl_xor(c, off, 32);
        }
        if (lane == 0) { redS[0] = a; redQ[0] = c; }
    }
    __syncthreads();
    const float mu   = redS[0] * (1.0f / DMODEL);
    const float var  = redQ[0] * (1.0f / DMODEL) - mu * mu;
    const float rstd = rsqrtf(var + 1e-6f);
#pragma unroll
    for (int i = 0; i < 4; ++i) {
        const int c = t + 256 * i;
        out[(size_t)row * DMODEL + c] = (v[i] - mu) * rstd * gamma[c] + beta[c];
    }
}

// ---------------------------------------------------------------------------
extern "C" void kernel_launch(void* const* d_in, const int* in_sizes, int n_in,
                              void* d_out, int out_size, void* d_ws, size_t ws_size,
                              hipStream_t stream) {
    const float* q     = (const float*)d_in[0];
    const float* k     = (const float*)d_in[1];
    const float* v     = (const float*)d_in[2];
    const int*   mask  = (const int*)  d_in[3];
    const float* mean  = (const float*)d_in[4];
    const float* stdv  = (const float*)d_in[5];
    const float* w_qs  = (const float*)d_in[6];
    const float* w_ks  = (const float*)d_in[7];
    const float* w_vs  = (const float*)d_in[8];
    const float* w_fc  = (const float*)d_in[9];
    const float* gamma = (const float*)d_in[10];
    const float* beta  = (const float*)d_in[11];
    float* out = (float*)d_out;

    // ---- workspace carve-up (256-B aligned)
    char* ws = (char*)d_ws;
    size_t off = 0;
    auto alloc = [&](size_t bytes) -> void* {
        void* p = ws + off;
        off += (bytes + 255) & ~(size_t)255;
        return p;
    };
    const size_t NTOKEL = (size_t)MTOK * DMODEL;          // 4096*1024
    __bf16* qbf  = (__bf16*)alloc(NTOKEL * 2);
    __bf16* kbf  = (__bf16*)alloc(NTOKEL * 2);
    __bf16* vbf  = (__bf16*)alloc(NTOKEL * 2);
    __bf16* wqsT = (__bf16*)alloc((size_t)DMODEL * DMODEL * 2);
    __bf16* wksT = (__bf16*)alloc((size_t)DMODEL * DMODEL * 2);
    __bf16* wvsT = (__bf16*)alloc((size_t)DMODEL * DMODEL * 2);
    __bf16* wfcT = (__bf16*)alloc((size_t)DMODEL * DMODEL * 2);
    __bf16* qhB  = (__bf16*)alloc(NTOKEL * 2);            // [B,H,LQ,DK]
    __bf16* khB  = (__bf16*)alloc(NTOKEL * 2);            // [B,H,LK,DK]
    __bf16* vhT  = (__bf16*)alloc(NTOKEL * 2);            // [B,H,DV,LK]
    __bf16* aout = (__bf16*)alloc(NTOKEL * 2);            // [B,LQ,H*DV]
    float*  proj = (float*) alloc(NTOKEL * 4);            // pre-LN f32

    // ---- phase 0: casts + weight transposes
    const int n4 = (int)(NTOKEL / 4);
    cast_bf16_kernel<<<(n4 + 255) / 256, 256, 0, stream>>>((const float4*)q, qbf, n4);
    cast_bf16_kernel<<<(n4 + 255) / 256, 256, 0, stream>>>((const float4*)k, kbf, n4);
    cast_bf16_kernel<<<(n4 + 255) / 256, 256, 0, stream>>>((const float4*)v, vbf, n4);
    dim3 tg(DMODEL / 32, DMODEL / 32);
    transpose_cast_kernel<<<tg, 256, 0, stream>>>(w_qs, wqsT, DMODEL, DMODEL);
    transpose_cast_kernel<<<tg, 256, 0, stream>>>(w_ks, wksT, DMODEL, DMODEL);
    transpose_cast_kernel<<<tg, 256, 0, stream>>>(w_vs, wvsT, DMODEL, DMODEL);
    transpose_cast_kernel<<<tg, 256, 0, stream>>>(w_fc, wfcT, DMODEL, DMODEL);

    // ---- phase 1: QKV projections (WMMA + async LDS staging)
    dim3 gg(DMODEL / 64, MTOK / 128);
    gemm_bf16_kernel<0><<<gg, 256, 0, stream>>>(qbf, wqsT, MTOK, DMODEL, DMODEL,
                                                qhB, nullptr, nullptr);
    gemm_bf16_kernel<0><<<gg, 256, 0, stream>>>(kbf, wksT, MTOK, DMODEL, DMODEL,
                                                khB, nullptr, nullptr);
    gemm_bf16_kernel<2><<<gg, 256, 0, stream>>>(vbf, wvsT, MTOK, DMODEL, DMODEL,
                                                vhT, nullptr, nullptr);

    // ---- phase 2: fused flash attention (WMMA + online softmax + async KV)
    dim3 ag(LQ / 64, NHEAD, BB);
    flash_attn_kernel<<<ag, 128, 0, stream>>>(qhB, khB, vhT, mask, mean, stdv, aout);

    // ---- phase 3: output projection + residual (WMMA), then LayerNorm
    gemm_bf16_kernel<3><<<gg, 256, 0, stream>>>(aout, wfcT, MTOK, DMODEL, DMODEL,
                                                nullptr, proj, q);
    layernorm_kernel<<<MTOK, 256, 0, stream>>>(proj, gamma, beta, out);
    (void)in_sizes; (void)n_in; (void)out_size; (void)ws_size;
}
